// RGCNSparseTIRNaiveLayer_58411555226290
// MI455X (gfx1250) — matrix operand order repaired and checked by reading, not compiled, not run on hardware
//
#include <hip/hip_runtime.h>
#include <hip/hip_bf16.h>

// ---------------------------------------------------------------------------
// RGCN forward on MI455X (gfx1250, wave32).
//
// Bucket edges by relation (padded to multiples of 32, sentinel pads map to
// src=-1-with-zero-A / dst=0-with-zero-add), then one 128-thread block
// (4 wave32s) per 32-edge group:
//   A0/A1 (16x32 bf16) = gathered feat rows, staged fp32->bf16 in LDS
//   B     (32x16 bf16) = W[r]^T tile (L2-resident W, shared by both A tiles)
//   D     (16x16 f32)  = v_wmma_f32_16x16x32_bf16, K=64 in 2 steps -> 4 wmma/wave
// then scatter-add the 32x64 tile into out[] with straight-line
// global_atomic_add_f32 (32-bit offsets, no branches).
// feat (51MB) and out (51MB) both fit the 192MB L2, so gather/scatter traffic
// stays on-chip; HBM sees ~edge lists only.
// ---------------------------------------------------------------------------

typedef __bf16 v16bf __attribute__((ext_vector_type(16)));
typedef float  v8f   __attribute__((ext_vector_type(8)));

#define F_DIM 64    // F_IN == F_OUT == 64
#define GRP   32    // edges per block

__device__ __forceinline__ __bf16 f2bf(float f) { return static_cast<__bf16>(f); }

union AFrag { struct { uint4 lo, hi; } q; v16bf v; };
union BFrag { __bf16 h[16]; v16bf v; };
union I4    { int4 q; int s[4]; };

// ---- phase 0: zero output, fill bucket with -1 sentinel, zero counters ----
__global__ void rgcn_init_kernel(float* __restrict__ out, long long outN,
                                 int* __restrict__ bucket, long long bucketN,
                                 int* __restrict__ counts, int R) {
    long long stride = (long long)gridDim.x * blockDim.x;
    long long total  = outN > bucketN ? outN : bucketN;
    for (long long i = (long long)blockIdx.x * blockDim.x + threadIdx.x;
         i < total; i += stride) {
        if (i < outN)    out[i]    = 0.0f;
        if (i < bucketN) bucket[i] = -1;
    }
    if (blockIdx.x == 0 && threadIdx.x < (unsigned)R) counts[threadIdx.x] = 0;
}

// ---- phase 1: per-relation histogram ----
__global__ void rgcn_hist_kernel(const int* __restrict__ etype, int E,
                                 int* __restrict__ counts) {
    int stride = gridDim.x * blockDim.x;
    for (int e = blockIdx.x * blockDim.x + threadIdx.x; e < E; e += stride)
        atomicAdd(&counts[etype[e]], 1);
}

// ---- phase 2: GRP-padded exclusive prefix sum over R (= 16) relations ----
__global__ void rgcn_offsets_kernel(const int* __restrict__ counts,
                                    int* __restrict__ paddedOff,
                                    int* __restrict__ cursor, int R) {
    if (threadIdx.x == 0 && blockIdx.x == 0) {
        int acc = 0;
        for (int r = 0; r < R; ++r) {
            paddedOff[r] = acc;
            cursor[r]    = acc;
            acc += ((counts[r] + (GRP - 1)) / GRP) * GRP;
        }
        paddedOff[R] = acc;                           // padded total
    }
}

// ---- phase 3: scatter edge ids into relation buckets ----
__global__ void rgcn_scatter_kernel(const int* __restrict__ etype, int E,
                                    int* __restrict__ cursor,
                                    int* __restrict__ bucket) {
    int stride = gridDim.x * blockDim.x;
    for (int e = blockIdx.x * blockDim.x + threadIdx.x; e < E; e += stride) {
        int pos = atomicAdd(&cursor[etype[e]], 1);
        bucket[pos] = e;
    }
}

// ---- phase 4: WMMA scatter-GEMM over 32-edge groups ----
__global__ void __launch_bounds__(128)
rgcn_wmma_kernel(const float* __restrict__ feat,
                 const float* __restrict__ W,      // (R, 64, 64) row-major
                 const int*   __restrict__ src,
                 const int*   __restrict__ dst,
                 const int*   __restrict__ bucket,
                 const int*   __restrict__ paddedOff,
                 int R,
                 float* __restrict__ out) {
    __shared__ __align__(16) int   s_dst[GRP];
    __shared__ int                 s_src[GRP];
    __shared__ __align__(16) __bf16 s_A[GRP * F_DIM];   // 32 gathered rows, bf16

    const int g    = blockIdx.x;
    const int base = g * GRP;
    if (base >= paddedOff[R]) return;                   // beyond padded total

    // Relation of this group (padded regions never straddle a group).
    int r = 0;
    for (int q = 0; q < R; ++q)
        if (base >= paddedOff[q]) r = q;

    const int tid = threadIdx.x;

    // Stage src/dst for the 32 edges. Pad rows: src=-1 (A row forced to zero
    // below) and dst=0 (their WMMA output is exactly +/-0 -> atomic no-op).
    if (tid < GRP) {
        int e = bucket[base + tid];
        s_src[tid] = (e >= 0) ? src[e] : -1;            // -1 marks pad for A stage
        s_dst[tid] = (e >= 0) ? dst[e] : 0;
    }
    __syncthreads();

    // Stage A = feat[src[0..31]] as bf16 into LDS: 128 thr x 16 floats.
    {
        int row = tid >> 2;            // 0..31
        int seg = (tid & 3) << 4;      // 0,16,32,48
        int srow = s_src[row];
        __bf16 h[16];
        if (srow >= 0) {
            const float* fp = feat + (long long)srow * F_DIM + seg;
            float4 a0 = *(const float4*)(fp);
            float4 a1 = *(const float4*)(fp + 4);
            float4 a2 = *(const float4*)(fp + 8);
            float4 a3 = *(const float4*)(fp + 12);
            h[0]  = f2bf(a0.x); h[1]  = f2bf(a0.y); h[2]  = f2bf(a0.z); h[3]  = f2bf(a0.w);
            h[4]  = f2bf(a1.x); h[5]  = f2bf(a1.y); h[6]  = f2bf(a1.z); h[7]  = f2bf(a1.w);
            h[8]  = f2bf(a2.x); h[9]  = f2bf(a2.y); h[10] = f2bf(a2.z); h[11] = f2bf(a2.w);
            h[12] = f2bf(a3.x); h[13] = f2bf(a3.y); h[14] = f2bf(a3.z); h[15] = f2bf(a3.w);
        } else {
            #pragma unroll
            for (int i = 0; i < 16; ++i) h[i] = __bf16(0.0f);
        }
        *(uint4*)&s_A[row * F_DIM + seg]     = *(const uint4*)&h[0];   // ds_store_b128
        *(uint4*)&s_A[row * F_DIM + seg + 8] = *(const uint4*)&h[8];
    }
    __syncthreads();

    const int lane  = tid & 31;
    const int wave  = tid >> 5;          // 0..3 -> output cols [wave*16, +16)
    const int col   = lane & 15;         // n within tile / m for A rows
    const int half  = lane >> 4;         // lane-group selector
    const int nbase = wave << 4;

    const float* Wr = W + (long long)r * (F_DIM * F_DIM)
                        + (long long)(nbase + col) * F_DIM;   // W row = out col

    v8f acc0 = {};   // edges 0..15
    v8f acc1 = {};   // edges 16..31

    #pragma unroll
    for (int kb = 0; kb < F_DIM; kb += 32) {
        // B fragment: column n = nbase+col; K = kb + half*16 + [0,16)
        BFrag bf;
        {
            const float* wp = Wr + kb + half * 16;
            float4 w0 = *(const float4*)(wp);
            float4 w1 = *(const float4*)(wp + 4);
            float4 w2 = *(const float4*)(wp + 8);
            float4 w3 = *(const float4*)(wp + 12);
            bf.h[0]  = f2bf(w0.x); bf.h[1]  = f2bf(w0.y); bf.h[2]  = f2bf(w0.z); bf.h[3]  = f2bf(w0.w);
            bf.h[4]  = f2bf(w1.x); bf.h[5]  = f2bf(w1.y); bf.h[6]  = f2bf(w1.z); bf.h[7]  = f2bf(w1.w);
            bf.h[8]  = f2bf(w2.x); bf.h[9]  = f2bf(w2.y); bf.h[10] = f2bf(w2.z); bf.h[11] = f2bf(w2.w);
            bf.h[12] = f2bf(w3.x); bf.h[13] = f2bf(w3.y); bf.h[14] = f2bf(w3.z); bf.h[15] = f2bf(w3.w);
        }
        // A fragments: row m = col (tile0) and m = col+16 (tile1);
        // K chunks [kb+half*16, +8) and [kb+half*16+8, +8)  (matches B: K = kb + half*16 + [0,16))
        AFrag a0, a1;
        a0.q.lo = *(const uint4*)&s_A[col * F_DIM + kb + half * 16];           // ds_load_b128
        a0.q.hi = *(const uint4*)&s_A[col * F_DIM + kb + half * 16 + 8];
        a1.q.lo = *(const uint4*)&s_A[(col + 16) * F_DIM + kb + half * 16];
        a1.q.hi = *(const uint4*)&s_A[(col + 16) * F_DIM + kb + half * 16 + 8];

        acc0 = __builtin_amdgcn_wmma_f32_16x16x32_bf16(
            false, a0.v, false, bf.v, (short)0, acc0, false, false);
        acc1 = __builtin_amdgcn_wmma_f32_16x16x32_bf16(
            false, a1.v, false, bf.v, (short)0, acc1, false, false);
    }

    // Preload the 16 dst rows this lane scatters to (contiguous -> b128 loads).
    int dlo[8], dhi[8];
    {
        I4 t;
        t.q = *(const int4*)&s_dst[half * 8];
        dlo[0] = t.s[0]; dlo[1] = t.s[1]; dlo[2] = t.s[2]; dlo[3] = t.s[3];
        t.q = *(const int4*)&s_dst[half * 8 + 4];
        dlo[4] = t.s[0]; dlo[5] = t.s[1]; dlo[6] = t.s[2]; dlo[7] = t.s[3];
        t.q = *(const int4*)&s_dst[16 + half * 8];
        dhi[0] = t.s[0]; dhi[1] = t.s[1]; dhi[2] = t.s[2]; dhi[3] = t.s[3];
        t.q = *(const int4*)&s_dst[16 + half * 8 + 4];
        dhi[4] = t.s[0]; dhi[5] = t.s[1]; dhi[6] = t.s[2]; dhi[7] = t.s[3];
    }

    // Scatter: lane holds D[m][n], n = nbase+col, m = v + 8*half (per tile).
    // Unconditional, 32-bit offsets -> straight-line global_atomic_add_f32.
    const unsigned n = (unsigned)(nbase + col);
    #pragma unroll
    for (int v = 0; v < 8; ++v)
        unsafeAtomicAdd(out + ((unsigned)dlo[v] * F_DIM + n), acc0[v]);
    #pragma unroll
    for (int v = 0; v < 8; ++v)
        unsafeAtomicAdd(out + ((unsigned)dhi[v] * F_DIM + n), acc1[v]);
}

// ---------------------------------------------------------------------------
extern "C" void kernel_launch(void* const* d_in, const int* in_sizes, int n_in,
                              void* d_out, int out_size, void* d_ws, size_t ws_size,
                              hipStream_t stream) {
    const float* feat  = (const float*)d_in[0];
    const float* W     = (const float*)d_in[1];
    const int*   src   = (const int*)d_in[2];
    const int*   dst   = (const int*)d_in[3];
    const int*   etype = (const int*)d_in[4];
    float*       out   = (float*)d_out;

    const int R = in_sizes[1] / (F_DIM * F_DIM);        // 16
    const int E = in_sizes[2];                           // 2,000,000
    const long long outN = (long long)out_size;          // 200,000 * 64

    // nGroups * GRP >= padded bucket total (each relation padded to mult of GRP)
    const int nGroups = (E + GRP - 1) / GRP + R;
    const long long bucketN = (long long)nGroups * GRP;

    // Workspace layout (~8.1 MB): counts | paddedOff | cursor | bucket
    int* counts    = (int*)d_ws;            // 32 ints
    int* paddedOff = (int*)d_ws + 32;       // R+1 ints
    int* cursor    = (int*)d_ws + 96;       // 32 ints
    int* bucket    = (int*)d_ws + 128;      // nGroups*GRP ints

    rgcn_init_kernel<<<2048, 256, 0, stream>>>(out, outN, bucket, bucketN, counts, R);
    rgcn_hist_kernel<<<(E + 255) / 256, 256, 0, stream>>>(etype, E, counts);
    rgcn_offsets_kernel<<<1, 32, 0, stream>>>(counts, paddedOff, cursor, R);
    rgcn_scatter_kernel<<<(E + 255) / 256, 256, 0, stream>>>(etype, E, cursor, bucket);
    rgcn_wmma_kernel<<<nGroups, 128, 0, stream>>>(feat, W, src, dst, bucket,
                                                  paddedOff, R, out);
}
